// AttentionHead_42984032698613
// MI455X (gfx1250) — matrix operand order repaired and checked by reading, not compiled, not run on hardware
//
#include <hip/hip_runtime.h>
#include <hip/hip_bf16.h>

// MI455X / gfx1250, wave32. All matrix math through v_wmma_f32_16x16x32_f16.
// Round 2: software-pipelined (double-buffered) inner loops so WMMA overlaps
// the HBM streams (partial s_wait_loadcnt instead of 0), exp2-domain softmax.
//
// Workspace layout (bytes):
//   x_f16   : B*S*I*2   = 33,554,432
//   Wq/Wk/Wv: H*I*2 * 3 =    393,216
//   Q/K/Vt  : B*S*H*2*3 =  6,291,456   (Vt stored (B,H,S) for contiguous PV B-frags)

typedef __attribute__((ext_vector_type(16))) _Float16 v16h;
typedef __attribute__((ext_vector_type(8)))  _Float16 v8h;
typedef __attribute__((ext_vector_type(8)))  float    v8f;

#define NB 8
#define SS 2048
#define II 1024
#define HH 64
#define NEG_INF_F (-1000000000.0f)

__device__ __forceinline__ v8f wmma_f16(v16h a, v16h b, v8f c) {
  // (neg_a, A, neg_b, B, c_mod, C, reuse_a, reuse_b)
  return __builtin_amdgcn_wmma_f32_16x16x32_f16(false, a, false, b, (short)0, c,
                                                false, false);
}

// A fragment (16x32 f16, wave32). rowptr = &src[row = tile_row0 + (lane&15)][k0].
// Lane<16 holds K {0..7, 16..23}; lane>=16 holds K {8..15, 24..31}.
__device__ __forceinline__ v16h load_a_frag(const _Float16* rowptr, int lane) {
  const int kb = (lane & 16) ? 8 : 0;
  union { v16h v; v8h h[2]; } u;
  u.h[0] = *(const v8h*)(rowptr + kb);
  u.h[1] = *(const v8h*)(rowptr + kb + 16);
  return u.v;
}

// B fragment (32x16 f16, wave32). Column n = lane&15 contiguous in memory.
// Lane<16 holds K {0..7, 16..23}; lane>=16 holds K {8..15, 24..31}.
__device__ __forceinline__ v16h load_b_frag(const _Float16* colptr, int lane) {
  const int kb = (lane & 16) ? 8 : 0;
  union { v16h v; v8h h[2]; } u;
  u.h[0] = *(const v8h*)(colptr + kb);
  u.h[1] = *(const v8h*)(colptr + kb + 16);
  return u.v;
}

// C-layout rows span 16 lanes (0..15 or 16..31); xor masks 1,2,4,8 stay in-half.
__device__ __forceinline__ float hrmax(float v) {
  v = fmaxf(v, __shfl_xor(v, 1, 32));
  v = fmaxf(v, __shfl_xor(v, 2, 32));
  v = fmaxf(v, __shfl_xor(v, 4, 32));
  v = fmaxf(v, __shfl_xor(v, 8, 32));
  return v;
}
__device__ __forceinline__ float hrsum(float v) {
  v += __shfl_xor(v, 1, 32);
  v += __shfl_xor(v, 2, 32);
  v += __shfl_xor(v, 4, 32);
  v += __shfl_xor(v, 8, 32);
  return v;
}

__global__ void cvt_f32_f16(const float* __restrict__ src,
                            _Float16* __restrict__ dst, long n) {
  long i = (long)blockIdx.x * blockDim.x + threadIdx.x;
  const long stride = (long)gridDim.x * blockDim.x;
  for (; i < n; i += stride) dst[i] = (_Float16)src[i];
}

// One wave = one 16-row tile of one of Q/K/V (blockIdx.y selects matrix).
// Double-buffered fragment loads: iteration k issues k+1's loads before its
// WMMAs so the waits on the WMMAs are partial.
__global__ __launch_bounds__(128) void qkv_proj(
    const _Float16* __restrict__ x,
    const _Float16* __restrict__ Wq, const _Float16* __restrict__ Wk,
    const _Float16* __restrict__ Wv,
    const float* __restrict__ bq, const float* __restrict__ bk,
    const float* __restrict__ bv,
    _Float16* __restrict__ Q, _Float16* __restrict__ K,
    _Float16* __restrict__ Vt) {
  const int lane = threadIdx.x & 31;
  const int wid  = threadIdx.x >> 5;
  const int rt   = blockIdx.x * 4 + wid;   // 0..1023 row tiles of 16
  const int sel  = blockIdx.y;             // 0=Q 1=K 2=V
  const int l16  = lane & 15;
  const int mrow = (lane & 16) ? 8 : 0;

  const _Float16* W    = (sel == 0) ? Wq : (sel == 1) ? Wk : Wv;
  const float*    bias = (sel == 0) ? bq : (sel == 1) ? bk : bv;

  const _Float16* arow = x + (size_t)(rt * 16 + l16) * II;
  const _Float16* wrow = W + (size_t)l16 * II;

  // prologue: k0 = 0 fragments
  v16h a  = load_a_frag(arow, lane);
  v16h b0 = load_b_frag(wrow, lane);
  v16h b1 = load_b_frag(wrow + (size_t)16 * II, lane);
  v16h b2 = load_b_frag(wrow + (size_t)32 * II, lane);
  v16h b3 = load_b_frag(wrow + (size_t)48 * II, lane);

  v8f c0 = {}, c1 = {}, c2 = {}, c3 = {};
  for (int k0 = 0; k0 < II; k0 += 32) {
    v16h an = a, b0n = b0, b1n = b1, b2n = b2, b3n = b3;
    const int kn = k0 + 32;
    if (kn < II) {  // issue next iteration's loads before consuming current
      an  = load_a_frag(arow + kn, lane);
      b0n = load_b_frag(wrow + kn, lane);
      b1n = load_b_frag(wrow + (size_t)16 * II + kn, lane);
      b2n = load_b_frag(wrow + (size_t)32 * II + kn, lane);
      b3n = load_b_frag(wrow + (size_t)48 * II + kn, lane);
    }
    c0 = wmma_f16(a, b0, c0);
    c1 = wmma_f16(a, b1, c1);
    c2 = wmma_f16(a, b2, c2);
    c3 = wmma_f16(a, b3, c3);
    a = an; b0 = b0n; b1 = b1n; b2 = b2n; b3 = b3n;
  }

  _Float16* dst = (sel == 0) ? Q : K;
  v8f cc[4] = {c0, c1, c2, c3};
#pragma unroll
  for (int n = 0; n < 4; ++n) {
    const int h = n * 16 + l16;
    const float bb = bias[h];
#pragma unroll
    for (int j = 0; j < 8; ++j) {
      const int grow = rt * 16 + mrow + j;
      const float val = cc[n][j] + bb;
      if (sel == 2) {
        const int bat = grow >> 11, srow = grow & (SS - 1);
        Vt[((size_t)bat * HH + h) * SS + srow] = (_Float16)val;  // (B,H,S)
      } else {
        dst[(size_t)grow * HH + h] = (_Float16)val;              // (B*S,H)
      }
    }
  }
}

// One wave = one 16-query tile; flash-style pass over 2048 keys in 32-chunks.
// K fragments + mask dwords are double-buffered one chunk ahead; V fragments
// are issued before the softmax VALU block so their latency is covered.
__global__ __launch_bounds__(128) void flash_attn(
    const _Float16* __restrict__ Qh, const _Float16* __restrict__ Kh,
    const _Float16* __restrict__ Vth, const int* __restrict__ mask,
    float* __restrict__ out) {
  __shared__ _Float16 ldsP[4][16 * 40];  // per-wave 16x32 prob tile, 80B rows
  const int lane = threadIdx.x & 31;
  const int wid  = threadIdx.x >> 5;
  const int tile = blockIdx.x * 4 + wid;  // 0..1023
  const int b    = tile >> 7;
  const int q0   = (tile & 127) * 16;
  const int l16  = lane & 15;
  const int mrow = (lane & 16) ? 8 : 0;
  // softmax in base-2 domain: fold 1/sqrt(64) * log2(e) into one constant
  const float cl2 = 0.125f * 1.44269504088896f;

  const _Float16* qrow = Qh + (size_t)(b * SS + q0 + l16) * HH;
  const v16h aq0 = load_a_frag(qrow, lane);
  const v16h aq1 = load_a_frag(qrow + 32, lane);

  v8f o0 = {}, o1 = {}, o2 = {}, o3 = {};
  float m[8], l[8];
#pragma unroll
  for (int j = 0; j < 8; ++j) { m[j] = NEG_INF_F; l[j] = 0.0f; }

  const int* mbase = mask + ((size_t)(b * SS + q0 + mrow)) * SS + l16;
  _Float16* slab = ldsP[wid];
  const _Float16* kbat = Kh + (size_t)b * SS * HH;
  const _Float16* vbat = Vth + ((size_t)b * HH + l16) * SS;

  // ---- prologue: chunk 0 K fragments + mask dwords
  v16h k00 = load_b_frag(kbat + (size_t)(0 + l16) * HH, lane);
  v16h k01 = load_b_frag(kbat + (size_t)(0 + l16) * HH + 32, lane);
  v16h k10 = load_b_frag(kbat + (size_t)(16 + l16) * HH, lane);
  v16h k11 = load_b_frag(kbat + (size_t)(16 + l16) * HH + 32, lane);
  int ma[8], mb[8];
#pragma unroll
  for (int j = 0; j < 8; ++j) {
    ma[j] = mbase[(size_t)j * SS];
    mb[j] = mbase[(size_t)j * SS + 16];
  }

  for (int kc = 0; kc < 64; ++kc) {
    const int k0 = kc * 32;

    // ---- issue next chunk's K fragments + mask loads (not yet waited on)
    v16h nk00 = k00, nk01 = k01, nk10 = k10, nk11 = k11;
    int na[8], nb[8];
#pragma unroll
    for (int j = 0; j < 8; ++j) { na[j] = ma[j]; nb[j] = mb[j]; }
    if (kc < 63) {
      const int n0 = k0 + 32;
      nk00 = load_b_frag(kbat + (size_t)(n0 + l16) * HH, lane);
      nk01 = load_b_frag(kbat + (size_t)(n0 + l16) * HH + 32, lane);
      nk10 = load_b_frag(kbat + (size_t)(n0 + 16 + l16) * HH, lane);
      nk11 = load_b_frag(kbat + (size_t)(n0 + 16 + l16) * HH + 32, lane);
      const int* mn = mbase + n0;
#pragma unroll
      for (int j = 0; j < 8; ++j) {
        na[j] = mn[(size_t)j * SS];
        nb[j] = mn[(size_t)j * SS + 16];
      }
      if (kc < 62) __builtin_prefetch(mbase + n0 + 32, 0, 1);  // 2 chunks ahead
    }

    // ---- scores S = Q K^T for 32 keys (two 16-key C tiles, 2 WMMAs each)
    v8f s0 = {}, s1 = {};
    s0 = wmma_f16(aq0, k00, s0);
    s0 = wmma_f16(aq1, k01, s0);
    s1 = wmma_f16(aq0, k10, s1);
    s1 = wmma_f16(aq1, k11, s1);

    // ---- issue this chunk's V fragments before the softmax VALU block
    const _Float16* vb = vbat + k0;
    const v16h v0 = load_b_frag(vb, lane);
    const v16h v1 = load_b_frag(vb + (size_t)16 * SS, lane);
    const v16h v2 = load_b_frag(vb + (size_t)32 * SS, lane);
    const v16h v3 = load_b_frag(vb + (size_t)48 * SS, lane);

    // ---- mask + online softmax (per C-row = per VGPR j, across 16-lane half)
    float p0[8], p1[8];
#pragma unroll
    for (int j = 0; j < 8; ++j) {
      const float t0 = ma[j] ? s0[j] * cl2 : NEG_INF_F;  // log2 domain
      const float t1 = mb[j] ? s1[j] * cl2 : NEG_INF_F;
      const float rm = hrmax(fmaxf(t0, t1));
      const float nm = fmaxf(m[j], rm);
      const float corr = exp2f(m[j] - nm);
      const float e0 = exp2f(t0 - nm);
      const float e1 = exp2f(t1 - nm);
      l[j] = l[j] * corr + hrsum(e0 + e1);
      m[j] = nm;
      p0[j] = e0; p1[j] = e1;
      o0[j] *= corr; o1[j] *= corr; o2[j] *= corr; o3[j] *= corr;
    }

    // ---- C-layout probs -> A-layout f16 via per-wave LDS round trip
#pragma unroll
    for (int j = 0; j < 8; ++j) {
      slab[(mrow + j) * 40 + l16]      = (_Float16)p0[j];
      slab[(mrow + j) * 40 + 16 + l16] = (_Float16)p1[j];
    }
    const int kb = (lane & 16) ? 8 : 0;
    union { v16h v; v8h h[2]; } pu;
    pu.h[0] = *(const v8h*)(slab + l16 * 40 + kb);        // ds_load_b128
    pu.h[1] = *(const v8h*)(slab + l16 * 40 + kb + 16);
    const v16h pa = pu.v;

    // ---- O += P V   (V^T is (B,H,S): key dim contiguous per hidden column)
    o0 = wmma_f16(pa, v0, o0);
    o1 = wmma_f16(pa, v1, o1);
    o2 = wmma_f16(pa, v2, o2);
    o3 = wmma_f16(pa, v3, o3);

    // rotate double buffers
    k00 = nk00; k01 = nk01; k10 = nk10; k11 = nk11;
#pragma unroll
    for (int j = 0; j < 8; ++j) { ma[j] = na[j]; mb[j] = nb[j]; }
  }

#pragma unroll
  for (int j = 0; j < 8; ++j) {
    const float inv = 1.0f / l[j];
    float* orow = out + (size_t)(b * SS + q0 + mrow + j) * HH;
    orow[l16]      = o0[j] * inv;
    orow[16 + l16] = o1[j] * inv;
    orow[32 + l16] = o2[j] * inv;
    orow[48 + l16] = o3[j] * inv;
  }
}

extern "C" void kernel_launch(void* const* d_in, const int* in_sizes, int n_in,
                              void* d_out, int out_size, void* d_ws,
                              size_t ws_size, hipStream_t stream) {
  (void)in_sizes; (void)n_in; (void)out_size; (void)ws_size;
  const float* input = (const float*)d_in[0];
  const int*   mask  = (const int*)d_in[1];
  const float* W_q   = (const float*)d_in[2];
  const float* b_q   = (const float*)d_in[3];
  const float* W_k   = (const float*)d_in[4];
  const float* b_k   = (const float*)d_in[5];
  const float* W_v   = (const float*)d_in[6];
  const float* b_v   = (const float*)d_in[7];
  float* out = (float*)d_out;

  char* ws = (char*)d_ws;
  size_t off = 0;
  _Float16* xh  = (_Float16*)(ws + off); off += (size_t)NB * SS * II * 2;
  _Float16* Wqh = (_Float16*)(ws + off); off += (size_t)HH * II * 2;
  _Float16* Wkh = (_Float16*)(ws + off); off += (size_t)HH * II * 2;
  _Float16* Wvh = (_Float16*)(ws + off); off += (size_t)HH * II * 2;
  _Float16* Qh  = (_Float16*)(ws + off); off += (size_t)NB * SS * HH * 2;
  _Float16* Kh  = (_Float16*)(ws + off); off += (size_t)NB * SS * HH * 2;
  _Float16* Vth = (_Float16*)(ws + off); off += (size_t)NB * SS * HH * 2;

  cvt_f32_f16<<<4096, 256, 0, stream>>>(input, xh, (long)NB * SS * II);
  cvt_f32_f16<<<256, 256, 0, stream>>>(W_q, Wqh, (long)HH * II);
  cvt_f32_f16<<<256, 256, 0, stream>>>(W_k, Wkh, (long)HH * II);
  cvt_f32_f16<<<256, 256, 0, stream>>>(W_v, Wvh, (long)HH * II);

  // 1024 row tiles / 4 waves per block, x3 matrices
  qkv_proj<<<dim3(256, 3), 128, 0, stream>>>(xh, Wqh, Wkh, Wvh, b_q, b_k, b_v,
                                             Qh, Kh, Vth);
  // 1024 q-tiles / 4 waves per block
  flash_attn<<<256, 128, 0, stream>>>(Qh, Kh, Vth, mask, out);
}
